// CfCActorCritic_22067541966961
// MI455X (gfx1250) — matrix operand order repaired and checked by reading, not compiled, not run on hardware
//
#include <hip/hip_runtime.h>

typedef __attribute__((ext_vector_type(16))) _Float16 v16h;
typedef __attribute__((ext_vector_type(8)))  _Float16 v8h;
typedef __attribute__((ext_vector_type(8)))  float    v8f;

#define B_   256
#define T_   1024
#define OBS_ 32
#define H_   128
#define A_   8

// LDS layout (offsets in _Float16 units, after the float bias region)
#define OFF_WB   0          // WbT:   128 x 160  (W^T, row n, stride 160)
#define OFF_W1   20480      // Wff1T: 128 x 128
#define OFF_W2   36864      // Wff2T: 128 x 128
#define OFF_W34  53248      // (Wta+Wtb)^T: 128 x 128
#define OFF_WH   69632      // WheadT: 16 x 128 (cols 0-7 = Wa, col 8 = Wc, rest 0)
#define OFF_Z    71680      // z: 16 x 160  [m][k] : k<32 obs, k>=32 hidden state h
#define OFF_ZB   74240      // zb: 16 x 128 (backbone activation)
#define HALVES_TOTAL 76288
#define FLOATS_TOTAL 528    // bb(128) b1(128) b2(128) b34(128) bhead(16)
#define SMEM_BYTES   (FLOATS_TOTAL*4 + HALVES_TOTAL*2)

#define L2E_F      1.4426950408889634f      // log2(e)
#define TANH_SC    2.8853900817779268f      // 2*log2(e)
#define BTANH_SC   1.9216697944641252f      // 0.666*2*log2(e)

// y = 2*log2(e)*x ; returns tanh(x). Uses v_exp_f32 + v_rcp_f32 (no IEEE div).
static __device__ __forceinline__ float tanh_exp2(float y) {
    y = fminf(fmaxf(y, -34.f), 34.f);
    float e = __builtin_amdgcn_exp2f(y);
    return (e - 1.f) * __builtin_amdgcn_rcpf(e + 1.f);
}
static __device__ __forceinline__ float sigm_fast(float x) {
    return __builtin_amdgcn_rcpf(1.f + __builtin_amdgcn_exp2f(-L2E_F * x));
}
static __device__ __forceinline__ v8h ld8(const _Float16* p) {
    return *(const v8h*)p;
}
static __device__ __forceinline__ v16h mk16(v8h lo, v8h hi2) {
    union { v16h v; v8h p[2]; } u;
    u.p[0] = lo; u.p[1] = hi2;
    return u.v;
}
static __device__ __forceinline__ v8f wmma16(v16h a, v16h b, v8f c) {
    return __builtin_amdgcn_wmma_f32_16x16x32_f16(false, a, false, b, (short)0, c,
                                                  false, false);
}

__global__ __launch_bounds__(256, 1)
void cfc_scan_fused(const float* __restrict__ obs,
                    const float* __restrict__ Wb,  const float* __restrict__ bb,
                    const float* __restrict__ W1,  const float* __restrict__ b1,
                    const float* __restrict__ W2,  const float* __restrict__ b2,
                    const float* __restrict__ W3,  const float* __restrict__ b3,
                    const float* __restrict__ W4,  const float* __restrict__ b4,
                    const float* __restrict__ Wa,  const float* __restrict__ ba,
                    const float* __restrict__ Wc,  const float* __restrict__ bc,
                    float* __restrict__ out)
{
    extern __shared__ char smem_raw[];
    float*    biasL = (float*)smem_raw;
    _Float16* hmem  = (_Float16*)(smem_raw + FLOATS_TOTAL * 4);

    const int tid  = threadIdx.x;
    const int lane = tid & 31;
    const int wv   = tid >> 5;       // wave id 0..7, owns N-tile columns n0..n0+15
    const int ln   = lane & 15;
    const int hi   = lane >> 4;
    const int b0   = blockIdx.x * 16;
    const int n0   = wv * 16;

    // ---- stage weights once: transpose to [n][k], convert f32 -> f16 ----
    for (int i = tid; i < 160 * 128; i += 256) {          // Wb [k=160][n=128]
        int n = i & 127, k = i >> 7;
        hmem[OFF_WB + n * 160 + k] = (_Float16)Wb[i];
    }
    for (int i = tid; i < 128 * 128; i += 256) {
        int n = i & 127, k = i >> 7;
        hmem[OFF_W1 + n * 128 + k] = (_Float16)W1[i];
    }
    for (int i = tid; i < 128 * 128; i += 256) {
        int n = i & 127, k = i >> 7;
        hmem[OFF_W2 + n * 128 + k] = (_Float16)W2[i];
    }
    for (int i = tid; i < 128 * 128; i += 256) {          // ts=1: fold ta+tb
        int n = i & 127, k = i >> 7;
        hmem[OFF_W34 + n * 128 + k] = (_Float16)(W3[i] + W4[i]);
    }
    for (int i = tid; i < 16 * 128; i += 256) {           // head: [Wa | Wc | 0]
        int n = i & 15, k = i >> 4;
        float v = 0.f;
        if (n < 8)       v = Wa[k * 8 + n];
        else if (n == 8) v = Wc[k];
        hmem[OFF_WH + n * 128 + k] = (_Float16)v;
    }
    for (int i = tid; i < FLOATS_TOTAL; i += 256) {
        float v;
        if (i < 128)      v = bb[i];
        else if (i < 256) v = b1[i - 128];
        else if (i < 384) v = b2[i - 256];
        else if (i < 512) v = b3[i - 384] + b4[i - 384];
        else { int a = i - 512; v = (a < 8) ? ba[a] : ((a == 8) ? bc[0] : 0.f); }
        biasL[i] = v;
    }
    // h_0 = 0
    for (int i = tid; i < 16 * 128; i += 256) {
        int m = i >> 7, k = i & 127;
        hmem[OFF_Z + m * 160 + 32 + k] = (_Float16)0.f;
    }
    // stage obs_0
    {
        int m = tid >> 5, k = tid & 31;
        hmem[OFF_Z + m * 160 + k] =
            (_Float16)obs[((size_t)(b0 + m) * T_) * OBS_ + k];
        hmem[OFF_Z + (m + 8) * 160 + k] =
            (_Float16)obs[((size_t)(b0 + m + 8) * T_) * OBS_ + k];
    }
    __syncthreads();

    // ---- hoist loop-invariant state into registers ----
    const float bvb  = biasL[n0 + ln];
    const float bv1  = biasL[128 + n0 + ln];
    const float bv2  = biasL[256 + n0 + ln];
    const float bv34 = biasL[384 + n0 + ln];
    const float bvh  = biasL[512 + ln];

    v16h wbF[5];
    #pragma unroll
    for (int kt = 0; kt < 5; ++kt) {
        const int k0 = kt * 32;
        wbF[kt] = mk16(ld8(&hmem[OFF_WB + (n0 + ln) * 160 + k0 + hi * 16]),
                       ld8(&hmem[OFF_WB + (n0 + ln) * 160 + k0 + hi * 16 + 8]));
    }
    v16h w1F[4], w2F[4], w34F[4];
    #pragma unroll
    for (int kt = 0; kt < 4; ++kt) {
        const int k0 = kt * 32;
        w1F[kt]  = mk16(ld8(&hmem[OFF_W1 + (n0 + ln) * 128 + k0 + hi * 16]),
                        ld8(&hmem[OFF_W1 + (n0 + ln) * 128 + k0 + hi * 16 + 8]));
        w2F[kt]  = mk16(ld8(&hmem[OFF_W2 + (n0 + ln) * 128 + k0 + hi * 16]),
                        ld8(&hmem[OFF_W2 + (n0 + ln) * 128 + k0 + hi * 16 + 8]));
        w34F[kt] = mk16(ld8(&hmem[OFF_W34 + (n0 + ln) * 128 + k0 + hi * 16]),
                        ld8(&hmem[OFF_W34 + (n0 + ln) * 128 + k0 + hi * 16 + 8]));
    }

    const int mo = tid >> 5;         // obs staging row (this row and row+8)
    const int ko = tid & 31;         // obs staging column
    const size_t VAL_OFF = (size_t)B_ * T_ * A_;

    // strength-reduced obs pointers for t+1 loads and t+2 prefetch
    const float* po0 = obs + ((size_t)(b0 + mo) * T_ + 1) * OBS_ + ko;
    const float* po1 = obs + ((size_t)(b0 + mo + 8) * T_ + 1) * OBS_ + ko;
    const float* pp0 = po0 + OBS_;
    const float* pp1 = po1 + OBS_;

    #pragma unroll 1
    for (int t = 0; t < T_; ++t) {
        // ---- early-issue obs_{t+1} global loads (overlap with compute) ----
        float o0 = 0.f, o1 = 0.f;
        if (t + 1 < T_) {
            o0 = *po0; o1 = *po1;
            po0 += OBS_; po1 += OBS_;
        }
        if (t + 2 < T_) {
            __builtin_prefetch(pp0, 0, 3);
            __builtin_prefetch(pp1, 0, 3);
            pp0 += OBS_; pp1 += OBS_;
        }

        // ---------- backbone: zb = 1.7159*tanh(0.666*([obs|h] @ Wb + bb)) ----------
        v8f accb;
        #pragma unroll
        for (int r = 0; r < 8; ++r) accb[r] = bvb;
        #pragma unroll
        for (int kt = 0; kt < 5; ++kt) {
            const int k0 = kt * 32;
            v16h a = mk16(ld8(&hmem[OFF_Z + ln * 160 + k0 + hi * 16]),
                          ld8(&hmem[OFF_Z + ln * 160 + k0 + hi * 16 + 8]));
            accb = wmma16(a, wbF[kt], accb);
        }
        #pragma unroll
        for (int r = 0; r < 8; ++r) {
            float v = 1.7159f * tanh_exp2(BTANH_SC * accb[r]);
            hmem[OFF_ZB + (r + 8 * hi) * 128 + n0 + ln] = (_Float16)v;
        }
        __syncthreads();

        // ---------- gate matmuls: ff1, ff2, (ta+tb) ----------
        v8f acc1, acc2, acc3;
        #pragma unroll
        for (int r = 0; r < 8; ++r) { acc1[r] = bv1; acc2[r] = bv2; acc3[r] = bv34; }
        #pragma unroll
        for (int kt = 0; kt < 4; ++kt) {
            const int k0 = kt * 32;
            v16h a = mk16(ld8(&hmem[OFF_ZB + ln * 128 + k0 + hi * 16]),
                          ld8(&hmem[OFF_ZB + ln * 128 + k0 + hi * 16 + 8]));
            acc1 = wmma16(a, w1F[kt], acc1);
            acc2 = wmma16(a, w2F[kt], acc2);
            acc3 = wmma16(a, w34F[kt], acc3);
        }
        // gate + write h_new back into z hidden region
        #pragma unroll
        for (int r = 0; r < 8; ++r) {
            float f1 = tanh_exp2(TANH_SC * acc1[r]);
            float f2 = tanh_exp2(TANH_SC * acc2[r]);
            float s  = sigm_fast(acc3[r]);             // sigmoid(zb@(Wta+Wtb)+bta+btb)
            float hn = f1 + s * (f2 - f1);             // ff1*(1-s) + s*ff2
            hmem[OFF_Z + (r + 8 * hi) * 160 + 32 + n0 + ln] = (_Float16)hn;
        }
        // store pre-loaded obs_{t+1} (obs region of z no longer read this step)
        hmem[OFF_Z + mo * 160 + ko]       = (_Float16)o0;
        hmem[OFF_Z + (mo + 8) * 160 + ko] = (_Float16)o1;
        __syncthreads();

        // ---------- actor/critic heads on h_new (rotating duty wave) ----------
        if (wv == (t & 7)) {
            v8f acch;
            #pragma unroll
            for (int r = 0; r < 8; ++r) acch[r] = bvh;
            #pragma unroll
            for (int kt = 0; kt < 4; ++kt) {
                const int k0 = kt * 32;
                v16h a = mk16(ld8(&hmem[OFF_Z + ln * 160 + 32 + k0 + hi * 16]),
                              ld8(&hmem[OFF_Z + ln * 160 + 32 + k0 + hi * 16 + 8]));
                v16h b = mk16(ld8(&hmem[OFF_WH + ln * 128 + k0 + hi * 16]),
                              ld8(&hmem[OFF_WH + ln * 128 + k0 + hi * 16 + 8]));
                acch = wmma16(a, b, acch);
            }
            #pragma unroll
            for (int r = 0; r < 8; ++r) {
                int brow = b0 + r + 8 * hi;            // D: M = r + 8*hi, N = ln
                if (ln < 8)
                    out[((size_t)brow * T_ + t) * A_ + ln] = acch[r];
                else if (ln == 8)
                    out[VAL_OFF + (size_t)brow * T_ + t] = acch[r];
            }
        }
    }
}

extern "C" void kernel_launch(void* const* d_in, const int* in_sizes, int n_in,
                              void* d_out, int out_size, void* d_ws, size_t ws_size,
                              hipStream_t stream) {
    (void)in_sizes; (void)n_in; (void)out_size; (void)d_ws; (void)ws_size;
    const float* obs = (const float*)d_in[0];
    const float* Wb  = (const float*)d_in[1];
    const float* bb  = (const float*)d_in[2];
    const float* W1  = (const float*)d_in[3];
    const float* b1  = (const float*)d_in[4];
    const float* W2  = (const float*)d_in[5];
    const float* b2  = (const float*)d_in[6];
    const float* W3  = (const float*)d_in[7];
    const float* b3  = (const float*)d_in[8];
    const float* W4  = (const float*)d_in[9];
    const float* b4  = (const float*)d_in[10];
    const float* Wa  = (const float*)d_in[11];
    const float* ba  = (const float*)d_in[12];
    const float* Wc  = (const float*)d_in[13];
    const float* bc  = (const float*)d_in[14];
    float* out = (float*)d_out;

    cfc_scan_fused<<<dim3(B_ / 16), dim3(256), SMEM_BYTES, stream>>>(
        obs, Wb, bb, W1, b1, W2, b2, W3, b3, W4, b4, Wa, ba, Wc, bc, out);
}
